// EncoderLayer_24068996727515
// MI455X (gfx1250) — matrix operand order repaired and checked by reading, not compiled, not run on hardware
//
#include <hip/hip_runtime.h>
#include <cstdint>
#include <cstddef>

// ---------------------------------------------------------------------------
// Types
// ---------------------------------------------------------------------------
typedef __bf16 bf16;
typedef __bf16  bf16x16 __attribute__((ext_vector_type(16)));
typedef float   v8f     __attribute__((ext_vector_type(8)));
typedef unsigned int u32x4 __attribute__((ext_vector_type(4)));
typedef int gvec4i __attribute__((vector_size(16)));   // builtin's pointee type

union Frag {            // one WMMA 16x32 bf16 operand = 32 bytes / lane
    bf16x16 v;
    u32x4   q[2];
};

#define WMMA_BF16(a, b, c) \
    __builtin_amdgcn_wmma_f32_16x16x32_bf16(false, (a), false, (b), (short)0, (c), false, false)

// ---- CDNA5 async global->LDS copy (ASYNCcnt) with portable fallback -------
#if __has_builtin(__builtin_amdgcn_global_load_async_to_lds_b128)
#define ASYNC_ON 1
#define ASYNC_B128(g, l)                                              \
    __builtin_amdgcn_global_load_async_to_lds_b128(                   \
        (__attribute__((address_space(1))) gvec4i*)(g),               \
        (__attribute__((address_space(3))) gvec4i*)(l), 0, 0)
#if __has_builtin(__builtin_amdgcn_s_wait_asynccnt)
#define WAIT_ASYNC() __builtin_amdgcn_s_wait_asynccnt(0)
#else
#define WAIT_ASYNC() asm volatile("s_wait_asynccnt 0x0" ::: "memory")
#endif
#else
#define ASYNC_ON 0
#define WAIT_ASYNC() ((void)0)
#endif

static constexpr int M_TOK   = 8192;   // 4 * 2048
static constexpr int D_MODEL = 1024;
static constexpr int D_FF    = 4096;
static constexpr int SEQ     = 2048;

// ---------------------------------------------------------------------------
// fp32 -> bf16 weight conversion (run once per launch; weights then stream
// into GEMM tiles via pure async b128 copies, no staging VALU).
// ---------------------------------------------------------------------------
__global__ __launch_bounds__(256)
void cvt_bf16_kernel(const float* __restrict__ w, bf16* __restrict__ out)
{
    int i = blockIdx.x * 1024 + threadIdx.x;
#pragma unroll
    for (int j = 0; j < 4; ++j, i += 256) out[i] = (bf16)w[i];
}

// ---------------------------------------------------------------------------
// LayerNorm (fp32 in -> bf16 out).  Bessel var, eps added to std (reference).
// ---------------------------------------------------------------------------
__global__ __launch_bounds__(256)
void ln_kernel(const float* __restrict__ x, const float* __restrict__ alpha,
               const float* __restrict__ beta, bf16* __restrict__ out)
{
    __shared__ float red[256];
    const int row = blockIdx.x;
    const int tid = threadIdx.x;
    const float* xr = x + (size_t)row * D_MODEL;

    float loc[4];
    float s = 0.f;
#pragma unroll
    for (int i = 0; i < 4; ++i) { loc[i] = xr[tid + i * 256]; s += loc[i]; }

    red[tid] = s; __syncthreads();
    for (int k = 128; k > 0; k >>= 1) { if (tid < k) red[tid] += red[tid + k]; __syncthreads(); }
    const float mu = red[0] * (1.0f / D_MODEL);
    __syncthreads();

    float sq = 0.f;
#pragma unroll
    for (int i = 0; i < 4; ++i) { float d = loc[i] - mu; sq += d * d; }
    red[tid] = sq; __syncthreads();
    for (int k = 128; k > 0; k >>= 1) { if (tid < k) red[tid] += red[tid + k]; __syncthreads(); }
    const float var = red[0] * (1.0f / (D_MODEL - 1));

    const float scale = alpha[0] / (sqrtf(var) + 1e-6f);
    const float shift = beta[0];

    bf16* orow = out + (size_t)row * D_MODEL;
#pragma unroll
    for (int i = 0; i < 4; ++i)
        orow[tid + i * 256] = (bf16)((loc[i] - mu) * scale + shift);
}

// ---------------------------------------------------------------------------
// GEMM:  C[M,N] = epi( A(bf16,[M,K]) @ Wb(bf16,[N,K])^T + bias )
// EPI 0: bf16 out (bias)                 — QKV projections
// EPI 1: bf16 out (bias + relu)          — FFN1
// EPI 2: f32 out (bias + residual add)   — O-proj / FFN2
// Tile 128x128x32, 256 threads = 8 waves; wave owns a 32x64 sub-tile.
// Double-buffered LDS; next tile staged with async global->LDS b128 copies
// overlapped with current tile's 8 WMMAs; one barrier per K-tile.
// ---------------------------------------------------------------------------
template<int EPI>
__global__ __launch_bounds__(256)
void gemm_bf16_kernel(const bf16* __restrict__ A, const bf16* __restrict__ Wb,
                      const float* __restrict__ bias, const float* __restrict__ res,
                      void* __restrict__ outp, int M, int N, int K)
{
    __shared__ __attribute__((aligned(16))) bf16 sA[2][128 * 32];
    __shared__ __attribute__((aligned(16))) bf16 sB[2][128 * 32];

    const int tid  = threadIdx.x;
    const int lane = tid & 31;
    const int wv   = tid >> 5;
    const int l15  = lane & 15;
    const int hi   = lane >> 4;
    const int wm   = wv >> 1;           // 0..3 : 32-row band
    const int wn   = wv & 1;            // 0..1 : 64-col band

    const int tr = tid >> 1;            // 0..127 tile row for staging
    const int tc = (tid & 1) << 4;      // 0 / 16
    const bf16* Ag = A  + (size_t)(blockIdx.y * 128 + tr) * K + tc;
    const bf16* Bg = Wb + (size_t)(blockIdx.x * 128 + tr) * K + tc;

    auto stage = [&](int buf, int kt) {
        bf16* a0 = &sA[buf][tr * 32 + tc];
        bf16* b0 = &sB[buf][tr * 32 + tc];
#if ASYNC_ON
        ASYNC_B128(Ag + kt,     a0);
        ASYNC_B128(Ag + kt + 8, a0 + 8);
        ASYNC_B128(Bg + kt,     b0);
        ASYNC_B128(Bg + kt + 8, b0 + 8);
#else
        *(u32x4*)a0       = *(const u32x4*)(Ag + kt);
        *(u32x4*)(a0 + 8) = *(const u32x4*)(Ag + kt + 8);
        *(u32x4*)b0       = *(const u32x4*)(Bg + kt);
        *(u32x4*)(b0 + 8) = *(const u32x4*)(Bg + kt + 8);
#endif
    };

    v8f acc[2][4] = {};

    const int nT = K >> 5;              // K / 32 tiles
    stage(0, 0);
    WAIT_ASYNC();
    __syncthreads();

    for (int i = 0; i < nT; ++i) {
        const int cur = i & 1;

        // ---- fragments of current tile (contiguous k order: k = 16*hi + s) ----
        Frag a[2], b[4];
#pragma unroll
        for (int mi = 0; mi < 2; ++mi) {
            const int r = (wm * 32 + mi * 16 + l15) * 32;
            a[mi].q[0] = *(const u32x4*)&sA[cur][r + hi * 16];
            a[mi].q[1] = *(const u32x4*)&sA[cur][r + hi * 16 + 8];
        }
#pragma unroll
        for (int ni = 0; ni < 4; ++ni) {
            const int r = (wn * 64 + ni * 16 + l15) * 32 + hi * 16;
            b[ni].q[0] = *(const u32x4*)&sB[cur][r];
            b[ni].q[1] = *(const u32x4*)&sB[cur][r + 8];
        }

        // ---- stage next tile (overlaps with WMMAs below) ----
        if (i + 1 < nT) stage(cur ^ 1, (i + 1) << 5);

#pragma unroll
        for (int mi = 0; mi < 2; ++mi)
#pragma unroll
            for (int ni = 0; ni < 4; ++ni)
                acc[mi][ni] = WMMA_BF16(a[mi].v, b[ni].v, acc[mi][ni]);

        WAIT_ASYNC();
        __syncthreads();
    }

    // ---- epilogue ----
#pragma unroll
    for (int mi = 0; mi < 2; ++mi) {
#pragma unroll
        for (int ni = 0; ni < 4; ++ni) {
            const int gm0 = blockIdx.y * 128 + wm * 32 + mi * 16 + 8 * hi;
            const int gn  = blockIdx.x * 128 + wn * 64 + ni * 16 + l15;
            const float bv = bias[gn];
#pragma unroll
            for (int r = 0; r < 8; ++r) {
                const size_t off = (size_t)(gm0 + r) * N + gn;
                float val = acc[mi][ni][r] + bv;
                if (EPI == 1) val = fmaxf(val, 0.0f);
                if (EPI <= 1) ((bf16*)outp)[off] = (bf16)val;
                else          ((float*)outp)[off] = val + res[off];
            }
        }
    }
}

// ---------------------------------------------------------------------------
// Attention: one block per (b,h, 32-query block). 256 threads = 8 waves.
// 32x2048 bf16 scores live in LDS (128KB of the 320KB WGP LDS), masked
// softmax in place, QK^T and P@V via WMMA.
// ---------------------------------------------------------------------------
__global__ __launch_bounds__(256)
void attn_kernel(const bf16* __restrict__ qg, const bf16* __restrict__ kg,
                 const bf16* __restrict__ vg, const int* __restrict__ mask,
                 bf16* __restrict__ og)
{
    extern __shared__ __attribute__((aligned(16))) char smem[];
    bf16* sQ  = (bf16*)smem;        // 32 x 64
    bf16* sKV = sQ + 32 * 64;       // 64 x 64 (K row-major, then V transposed)
    bf16* sS  = sKV + 64 * 64;      // 32 x 2048 scores / probabilities

    __shared__ float red[256];
    __shared__ float rowstat[32];

    const int bh = blockIdx.x;            // 0..63
    const int b  = bh >> 4;
    const int h  = bh & 15;
    const int qb = blockIdx.y;            // 0..63 (32 queries each)

    const int tid  = threadIdx.x;
    const int lane = tid & 31;
    const int wv   = tid >> 5;
    const int l15  = lane & 15;
    const int hi   = lane >> 4;
    const int mi   = wv >> 2;             // 0..1 : query 16-row band
    const int ni   = wv & 3;              // 0..3 : 16-col band

    // ---- load Q tile (32x64) ----
    {
        const int idx = tid * 8;
        const int r = idx >> 6, c = idx & 63;
        const bf16* src = qg + (size_t)(b * SEQ + qb * 32 + r) * D_MODEL + h * 64 + c;
#if ASYNC_ON
        ASYNC_B128(src, &sQ[r * 64 + c]);
#else
        *(u32x4*)&sQ[r * 64 + c] = *(const u32x4*)src;
#endif
    }

    // ---- phase 1: S = (QK^T)*scale, masked, into sS ----
    const int idx2 = tid * 16;
    const int r2 = idx2 >> 6, c2 = idx2 & 63;

    for (int kc = 0; kc < 32; ++kc) {
        __syncthreads();
        const bf16* src = kg + (size_t)(b * SEQ + kc * 64 + r2) * D_MODEL + h * 64 + c2;
#if ASYNC_ON
        ASYNC_B128(src,     &sKV[r2 * 64 + c2]);
        ASYNC_B128(src + 8, &sKV[r2 * 64 + c2 + 8]);
        WAIT_ASYNC();
#else
        *(u32x4*)&sKV[r2 * 64 + c2]     = *(const u32x4*)src;
        *(u32x4*)&sKV[r2 * 64 + c2 + 8] = *(const u32x4*)(src + 8);
#endif
        __syncthreads();

        v8f s = {};
#pragma unroll
        for (int ks = 0; ks < 2; ++ks) {
            Frag a, bfr;
            const int ar = (mi * 16 + l15) * 64 + ks * 32;
            a.q[0] = *(const u32x4*)&sQ[ar + hi * 16];
            a.q[1] = *(const u32x4*)&sQ[ar + hi * 16 + 8];
            const int br = (ni * 16 + l15) * 64 + ks * 32 + hi * 16;
            bfr.q[0] = *(const u32x4*)&sKV[br];
            bfr.q[1] = *(const u32x4*)&sKV[br + 8];
            s = WMMA_BF16(a.v, bfr.v, s);
        }
        const int tg = kc * 64 + ni * 16 + l15;              // key index
        const int mv = mask[b * SEQ + tg];
#pragma unroll
        for (int r = 0; r < 8; ++r) {
            const float val = mv ? s[r] * 0.125f : -1e9f;
            sS[(mi * 16 + r + 8 * hi) * SEQ + tg] = (bf16)val;
        }
    }

    // ---- phase 2: softmax over each of the 32 rows (8 threads / row) ----
    __syncthreads();
    {
        const int r   = tid >> 3;
        const int seg = tid & 7;
        bf16* rowp = &sS[r * SEQ + seg * 256];

        float mx = -3.0e38f;
        for (int i = 0; i < 256; ++i) mx = fmaxf(mx, (float)rowp[i]);
        red[tid] = mx; __syncthreads();
        if (seg == 0) {
            float m2 = red[tid];
#pragma unroll
            for (int j = 1; j < 8; ++j) m2 = fmaxf(m2, red[tid + j]);
            rowstat[r] = m2;
        }
        __syncthreads();
        const float rmax = rowstat[r];

        float sum = 0.f;
        for (int i = 0; i < 256; ++i) {
            const float e = __expf((float)rowp[i] - rmax);
            rowp[i] = (bf16)e;
            sum += e;
        }
        red[tid] = sum; __syncthreads();
        if (seg == 0) {
            float s2 = red[tid];
#pragma unroll
            for (int j = 1; j < 8; ++j) s2 += red[tid + j];
            rowstat[r] = s2;
        }
        __syncthreads();
        const float inv = 1.0f / rowstat[r];
        for (int i = 0; i < 256; ++i) rowp[i] = (bf16)((float)rowp[i] * inv);
    }

    // ---- phase 3: O = P @ V ----
    v8f o = {};
    for (int vc = 0; vc < 32; ++vc) {
        __syncthreads();
        // load V chunk transposed: sKV[d][t]
        const bf16* src = vg + (size_t)(b * SEQ + vc * 64 + r2) * D_MODEL + h * 64 + c2;
        union { u32x4 q[2]; bf16 e[16]; } vb;
        vb.q[0] = *(const u32x4*)src;
        vb.q[1] = *(const u32x4*)(src + 8);
#pragma unroll
        for (int j = 0; j < 16; ++j) sKV[(c2 + j) * 64 + r2] = vb.e[j];
        __syncthreads();

#pragma unroll
        for (int ks = 0; ks < 2; ++ks) {
            Frag a, bfr;
            const int ar = (mi * 16 + l15) * SEQ + vc * 64 + ks * 32;
            a.q[0] = *(const u32x4*)&sS[ar + hi * 16];
            a.q[1] = *(const u32x4*)&sS[ar + hi * 16 + 8];
            const int br = (ni * 16 + l15) * 64 + ks * 32 + hi * 16;  // sKV[d][t]
            bfr.q[0] = *(const u32x4*)&sKV[br];
            bfr.q[1] = *(const u32x4*)&sKV[br + 8];
            o = WMMA_BF16(a.v, bfr.v, o);
        }
    }

    // ---- write O (bf16, model layout) ----
#pragma unroll
    for (int r = 0; r < 8; ++r) {
        const int gm = qb * 32 + mi * 16 + r + 8 * hi;
        const int gc = h * 64 + ni * 16 + l15;
        og[(size_t)(b * SEQ + gm) * D_MODEL + gc] = (bf16)o[r];
    }
}

// ---------------------------------------------------------------------------
// Host-side orchestration
// ---------------------------------------------------------------------------
extern "C" void kernel_launch(void* const* d_in, const int* in_sizes, int n_in,
                              void* d_out, int out_size, void* d_ws, size_t ws_size,
                              hipStream_t stream)
{
    (void)in_sizes; (void)n_in; (void)out_size; (void)ws_size;

    const float* x     = (const float*)d_in[0];
    const int*   mask  = (const int*)d_in[1];
    const float* wq    = (const float*)d_in[2];
    const float* bq    = (const float*)d_in[3];
    const float* wk    = (const float*)d_in[4];
    const float* bk    = (const float*)d_in[5];
    const float* wv    = (const float*)d_in[6];
    const float* bv    = (const float*)d_in[7];
    const float* wo    = (const float*)d_in[8];
    const float* bo    = (const float*)d_in[9];
    const float* w1    = (const float*)d_in[10];
    const float* b1    = (const float*)d_in[11];
    const float* w2    = (const float*)d_in[12];
    const float* b2    = (const float*)d_in[13];
    const float* ln1_a = (const float*)d_in[14];
    const float* ln1_b = (const float*)d_in[15];
    const float* ln2_a = (const float*)d_in[16];
    const float* ln2_b = (const float*)d_in[17];

    // workspace layout
    char* ws = (char*)d_ws;
    const size_t SZ_BF_MD = (size_t)M_TOK * D_MODEL * sizeof(bf16);     // 16 MiB
    const size_t SZ_W_DD  = (size_t)D_MODEL * D_MODEL * sizeof(bf16);   //  2 MiB
    const size_t SZ_W_FD  = (size_t)D_FF * D_MODEL * sizeof(bf16);      //  8 MiB
    bf16*  hbf  = (bf16*)ws;   ws += SZ_BF_MD;    // LN1 out
    bf16*  qb_  = (bf16*)ws;   ws += SZ_BF_MD;
    bf16*  kb_  = (bf16*)ws;   ws += SZ_BF_MD;
    bf16*  vb_  = (bf16*)ws;   ws += SZ_BF_MD;
    bf16*  attn = (bf16*)ws;   ws += SZ_BF_MD;
    float* x1   = (float*)ws;  ws += (size_t)M_TOK * D_MODEL * sizeof(float);
    bf16*  h2   = (bf16*)ws;   ws += SZ_BF_MD;    // LN2 out
    bf16*  ffn1 = (bf16*)ws;   ws += (size_t)M_TOK * D_FF * sizeof(bf16);
    bf16*  wqb  = (bf16*)ws;   ws += SZ_W_DD;
    bf16*  wkb  = (bf16*)ws;   ws += SZ_W_DD;
    bf16*  wvb  = (bf16*)ws;   ws += SZ_W_DD;
    bf16*  wob  = (bf16*)ws;   ws += SZ_W_DD;
    bf16*  w1b  = (bf16*)ws;   ws += SZ_W_FD;
    bf16*  w2b  = (bf16*)ws;

    const dim3 blk(256);
    const dim3 gD(D_MODEL / 128, M_TOK / 128);     // (8, 64)
    const dim3 gF(D_FF / 128,    M_TOK / 128);     // (32, 64)
    const int  gW_DD = (D_MODEL * D_MODEL) / 1024; // 1024 blocks
    const int  gW_FD = (D_FF * D_MODEL) / 1024;    // 4096 blocks

    const size_t attn_smem = (32 * 64 + 64 * 64 + 32 * SEQ) * sizeof(bf16); // 143360 B
    (void)hipFuncSetAttribute((const void*)attn_kernel,
                              hipFuncAttributeMaxDynamicSharedMemorySize,
                              (int)attn_smem);

    // 0. weight fp32 -> bf16 (once per launch; tiny vs 23.3 TB/s)
    cvt_bf16_kernel<<<gW_DD, blk, 0, stream>>>(wq, wqb);
    cvt_bf16_kernel<<<gW_DD, blk, 0, stream>>>(wk, wkb);
    cvt_bf16_kernel<<<gW_DD, blk, 0, stream>>>(wv, wvb);
    cvt_bf16_kernel<<<gW_DD, blk, 0, stream>>>(wo, wob);
    cvt_bf16_kernel<<<gW_FD, blk, 0, stream>>>(w1, w1b);
    cvt_bf16_kernel<<<gW_FD, blk, 0, stream>>>(w2, w2b);

    // 1. LN1
    ln_kernel<<<M_TOK, blk, 0, stream>>>(x, ln1_a, ln1_b, hbf);

    // 2. Q/K/V projections (bf16 out)
    gemm_bf16_kernel<0><<<gD, blk, 0, stream>>>(hbf, wqb, bq, nullptr, qb_, M_TOK, D_MODEL, D_MODEL);
    gemm_bf16_kernel<0><<<gD, blk, 0, stream>>>(hbf, wkb, bk, nullptr, kb_, M_TOK, D_MODEL, D_MODEL);
    gemm_bf16_kernel<0><<<gD, blk, 0, stream>>>(hbf, wvb, bv, nullptr, vb_, M_TOK, D_MODEL, D_MODEL);

    // 3. attention core
    attn_kernel<<<dim3(64, 64), blk, attn_smem, stream>>>(qb_, kb_, vb_, mask, attn);

    // 4. output projection + residual -> x1 (f32)
    gemm_bf16_kernel<2><<<gD, blk, 0, stream>>>(attn, wob, bo, x, (void*)x1, M_TOK, D_MODEL, D_MODEL);

    // 5. LN2
    ln_kernel<<<M_TOK, blk, 0, stream>>>(x1, ln2_a, ln2_b, h2);

    // 6. FFN1 (relu, bf16 out)
    gemm_bf16_kernel<1><<<gF, blk, 0, stream>>>(h2, w1b, b1, nullptr, ffn1, M_TOK, D_FF, D_MODEL);

    // 7. FFN2 + residual -> d_out (f32)
    gemm_bf16_kernel<2><<<gD, blk, 0, stream>>>(ffn1, w2b, b2, x1, d_out, M_TOK, D_MODEL, D_FF);
}
